// MultiHeadAttention_1881195676420
// MI455X (gfx1250) — matrix operand order repaired and checked by reading, not compiled, not run on hardware
//
#include <hip/hip_runtime.h>
#include <hip/hip_bf16.h>
#include <stdint.h>

#define DMODEL 1024
#define NHEADS 16
#define DK     64
#define NB     2
#define SEQ    2048
#define NROWS  (NB * SEQ)   // 4096

typedef __attribute__((ext_vector_type(16))) __bf16 bf16x16;
typedef __attribute__((ext_vector_type(8)))  __bf16 bf16x8;
typedef __attribute__((ext_vector_type(8)))  float  f32x8;
typedef int v4i __attribute__((__vector_size__(16)));

#define AS1 __attribute__((address_space(1)))
#define AS3 __attribute__((address_space(3)))

// ---------------------------------------------------------------- WMMA helper
__device__ __forceinline__ f32x8 wmma_bf16f32(bf16x16 a, bf16x16 b, f32x8 c) {
  return __builtin_amdgcn_wmma_f32_16x16x32_bf16(false, a, false, b, (short)0, c,
                                                 false, false);
}

// A-fragment (16x32, row-major source, row stride ld)
__device__ __forceinline__ bf16x16 load_a_frag(const __bf16* base, int ld, int lr, int half) {
  const __bf16* p = base + lr * ld + half * 8;
  bf16x8 lo = *(const bf16x8*)(p);
  bf16x8 hi = *(const bf16x8*)(p + 16);
  return __builtin_shufflevector(lo, hi, 0, 1, 2, 3, 4, 5, 6, 7,
                                         8, 9, 10, 11, 12, 13, 14, 15);
}

// B-fragment for B[k][n] = R[n][k] (R row-major, row stride ld): k = half*8 + [0,8) and 16 + half*8 + [0,8)
__device__ __forceinline__ bf16x16 load_bt_frag(const __bf16* base, int ld, int lr, int half) {
  const __bf16* p = base + lr * ld + half * 8;
  bf16x8 lo = *(const bf16x8*)(p);
  bf16x8 hi = *(const bf16x8*)(p + 16);
  return __builtin_shufflevector(lo, hi, 0, 1, 2, 3, 4, 5, 6, 7,
                                         8, 9, 10, 11, 12, 13, 14, 15);
}

__device__ __forceinline__ float hmax16(float v) {
  v = fmaxf(v, __shfl_xor(v, 1, 32));
  v = fmaxf(v, __shfl_xor(v, 2, 32));
  v = fmaxf(v, __shfl_xor(v, 4, 32));
  v = fmaxf(v, __shfl_xor(v, 8, 32));
  return v;
}
__device__ __forceinline__ float hsum16(float v) {
  v += __shfl_xor(v, 1, 32);
  v += __shfl_xor(v, 2, 32);
  v += __shfl_xor(v, 4, 32);
  v += __shfl_xor(v, 8, 32);
  return v;
}

// ------------------------------------------------- async global->LDS copy (16B/lane)
__device__ __forceinline__ void async_copy16(const void* g, void* l) {
#if __has_builtin(__builtin_amdgcn_global_load_async_to_lds_b128)
  AS1 v4i* gp = (AS1 v4i*)(uintptr_t)g;
  AS3 v4i* lp = (AS3 v4i*)(unsigned int)(uintptr_t)l;
  __builtin_amdgcn_global_load_async_to_lds_b128(gp, lp, 0, 0);
#else
  *(uint4*)l = *(const uint4*)g;
#endif
}

__device__ __forceinline__ void async_wait0() {
#if __has_builtin(__builtin_amdgcn_s_wait_asynccnt)
  __builtin_amdgcn_s_wait_asynccnt(0);
#else
  asm volatile("s_wait_asynccnt 0" ::: "memory");
#endif
}

// ---------------------------------------------------------------- fp32 -> bf16
__global__ __launch_bounds__(256) void cvt_f32_bf16(const float* __restrict__ in,
                                                    __bf16* __restrict__ out, int n) {
  int i = blockIdx.x * 256 + threadIdx.x;
  if (i < n) out[i] = (__bf16)in[i];
}

// ---------------------------------------------------------------- QKV projection
// out = X @ W^T + b ; one wave per 64x64 output tile, blockIdx.z selects Q/K/V.
__global__ __launch_bounds__(32)
void qkv_proj_kernel(const __bf16* __restrict__ Xq, const __bf16* __restrict__ Xk,
                     const __bf16* __restrict__ Xv, const __bf16* __restrict__ Wq,
                     const __bf16* __restrict__ Wk, const __bf16* __restrict__ Wv,
                     const float* __restrict__ bq, const float* __restrict__ bk,
                     const float* __restrict__ bv, float* __restrict__ resid,
                     __bf16* __restrict__ Qh, __bf16* __restrict__ Kh,
                     __bf16* __restrict__ Vt) {
  const int lane = threadIdx.x;
  const int lr = lane & 15, half = lane >> 4;
  const int m0 = blockIdx.x * 64;
  const int n0 = blockIdx.y * 64;
  const int which = blockIdx.z;

  const __bf16* X = (which == 0) ? Xq : (which == 1) ? Xk : Xv;
  const __bf16* W = (which == 0) ? Wq : (which == 1) ? Wk : Wv;
  const float* bias = (which == 0) ? bq : (which == 1) ? bk : bv;

  f32x8 c[4][4] = {};
  for (int k0 = 0; k0 < DMODEL; k0 += 32) {
    bf16x16 a[4];
#pragma unroll
    for (int mi = 0; mi < 4; ++mi)
      a[mi] = load_a_frag(X + (m0 + mi * 16) * DMODEL + k0, DMODEL, lr, half);
#pragma unroll
    for (int ni = 0; ni < 4; ++ni) {
      bf16x16 bfr = load_bt_frag(W + (n0 + ni * 16) * DMODEL + k0, DMODEL, lr, half);
#pragma unroll
      for (int mi = 0; mi < 4; ++mi) c[mi][ni] = wmma_bf16f32(a[mi], bfr, c[mi][ni]);
    }
  }

#pragma unroll
  for (int mi = 0; mi < 4; ++mi)
#pragma unroll
    for (int ni = 0; ni < 4; ++ni) {
      const int n = n0 + ni * 16 + lr;
      const float bn = bias[n];
      const int h = n >> 6, d = n & 63;
#pragma unroll
      for (int r = 0; r < 8; ++r) {
        const int m = m0 + mi * 16 + half * 8 + r;  // C/D layout row
        const int b = m >> 11, s = m & (SEQ - 1);
        const float v = c[mi][ni][r] + bn;
        if (which == 0) {
          resid[m * DMODEL + n] = v;
          Qh[(((b * NHEADS) + h) * SEQ + s) * DK + d] = (__bf16)v;
        } else if (which == 1) {
          Kh[(((b * NHEADS) + h) * SEQ + s) * DK + d] = (__bf16)v;
        } else {
          Vt[(((b * NHEADS) + h) * DK + d) * SEQ + s] = (__bf16)v;
        }
      }
    }
}

// ---------------------------------------------------------------- flash attention
// 4 waves/block; block covers 64 q-rows of one (b,h). K/V chunks (32 keys) are
// double-buffered in LDS via async global->LDS copies shared by all 4 waves.
#define KLDS (32 * 72)  // 32 key rows, stride 72 elems (144B)
#define VLDS (64 * 40)  // 64 d rows,  stride 40 elems (80B)

__device__ __forceinline__ void stage_chunk(const __bf16* __restrict__ Kbase,
                                            const __bf16* __restrict__ Vbase,
                                            int kk0, __bf16* kbuf, __bf16* vbuf,
                                            int tid) {
#pragma unroll
  for (int j = 0; j < 2; ++j) {  // K: 32 rows x 64 elems = 256 16B units
    const int u = tid + j * 128;
    const int row = u >> 3, cu = u & 7;
    async_copy16(Kbase + (kk0 + row) * DK + cu * 8, kbuf + row * 72 + cu * 8);
  }
#pragma unroll
  for (int j = 0; j < 2; ++j) {  // V^T: 64 rows x 32 elems = 256 16B units
    const int u = tid + j * 128;
    const int d = u >> 2, cu = u & 3;
    async_copy16(Vbase + d * SEQ + kk0 + cu * 8, vbuf + d * 40 + cu * 8);
  }
}

__global__ __launch_bounds__(128)
void attn_kernel(const __bf16* __restrict__ Qh, const __bf16* __restrict__ Kh,
                 const __bf16* __restrict__ Vt, __bf16* __restrict__ ctxbuf) {
  __shared__ __align__(16) __bf16 Kld[2][KLDS];
  __shared__ __align__(16) __bf16 Vld[2][VLDS];
  __shared__ __align__(16) __bf16 plds[4][16 * 40];

  const int tid = threadIdx.x;
  const int wave = tid >> 5;
  const int lane = tid & 31;
  const int lr = lane & 15, half = lane >> 4;
  const int q0 = blockIdx.x * 64 + wave * 16;
  const int bh = blockIdx.y;  // b*16 + h
  const int b = bh >> 4, h = bh & 15;

  const __bf16* Qbase = Qh + (size_t)bh * SEQ * DK;  // [s][d]
  const __bf16* Kbase = Kh + (size_t)bh * SEQ * DK;  // [s][d]
  const __bf16* Vbase = Vt + (size_t)bh * DK * SEQ;  // [d][s]

  bf16x16 qa0 = load_a_frag(Qbase + q0 * DK + 0, DK, lr, half);
  bf16x16 qa1 = load_a_frag(Qbase + q0 * DK + 32, DK, lr, half);

  f32x8 ctx[4] = {};
  f32x8 mrow, lrow;
#pragma unroll
  for (int r = 0; r < 8; ++r) { mrow[r] = -1e30f; lrow[r] = 0.0f; }

  stage_chunk(Kbase, Vbase, 0, &Kld[0][0], &Vld[0][0], tid);
  async_wait0();
  __syncthreads();

  for (int kk0 = 0; kk0 < SEQ; kk0 += 32) {
    const int cur = (kk0 >> 5) & 1;
    if (kk0 + 32 < SEQ)
      stage_chunk(Kbase, Vbase, kk0 + 32, &Kld[cur ^ 1][0], &Vld[cur ^ 1][0], tid);

    const __bf16* kbuf = &Kld[cur][0];
    const __bf16* vbuf = &Vld[cur][0];

    // scores for 32 keys (two 16-key C tiles), K fragments from LDS
    f32x8 s0 = {}, s1 = {};
    {
      bf16x16 kb = load_bt_frag(kbuf + 0 * 72 + 0, 72, lr, half);
      s0 = wmma_bf16f32(qa0, kb, s0);
      kb = load_bt_frag(kbuf + 0 * 72 + 32, 72, lr, half);
      s0 = wmma_bf16f32(qa1, kb, s0);
      kb = load_bt_frag(kbuf + 16 * 72 + 0, 72, lr, half);
      s1 = wmma_bf16f32(qa0, kb, s1);
      kb = load_bt_frag(kbuf + 16 * 72 + 32, 72, lr, half);
      s1 = wmma_bf16f32(qa1, kb, s1);
    }

    // online softmax update (row r of this half <-> accumulator slot r)
    f32x8 p0, p1;
#pragma unroll
    for (int r = 0; r < 8; ++r) {
      const float x0 = s0[r] * 0.125f;  // 1/sqrt(64)
      const float x1 = s1[r] * 0.125f;
      const float cm = hmax16(fmaxf(x0, x1));
      const float mn = fmaxf(mrow[r], cm);
      const float al = __expf(mrow[r] - mn);
      const float e0 = __expf(x0 - mn);
      const float e1 = __expf(x1 - mn);
      lrow[r] = lrow[r] * al + hsum16(e0 + e1);
      mrow[r] = mn;
      p0[r] = e0;
      p1[r] = e1;
#pragma unroll
      for (int dt = 0; dt < 4; ++dt) ctx[dt][r] *= al;
    }

    // re-layout P (C/D layout) -> A-fragment layout through per-wave LDS tile
    __bf16* pw = &plds[wave][0];
#pragma unroll
    for (int r = 0; r < 8; ++r) {
      pw[(half * 8 + r) * 40 + lr] = (__bf16)p0[r];
      pw[(half * 8 + r) * 40 + 16 + lr] = (__bf16)p1[r];
    }
    asm volatile("s_wait_dscnt 0" ::: "memory");
    bf16x8 plo = *(const bf16x8*)(pw + lr * 40 + half * 8);
    bf16x8 phi = *(const bf16x8*)(pw + lr * 40 + 16 + half * 8);
    bf16x16 pa = __builtin_shufflevector(plo, phi, 0, 1, 2, 3, 4, 5, 6, 7,
                                         8, 9, 10, 11, 12, 13, 14, 15);

    // ctx += P @ V : V^T fragments from LDS
#pragma unroll
    for (int dt = 0; dt < 4; ++dt) {
      bf16x16 vb = load_bt_frag(vbuf + (dt * 16) * 40, 40, lr, half);
      ctx[dt] = wmma_bf16f32(pa, vb, ctx[dt]);
    }

    async_wait0();   // next chunk resident before anyone flips buffers
    __syncthreads(); // also orders buffer reuse across waves
  }

  // normalize and store ctx -> [b*S+s][h*64+d] bf16
#pragma unroll
  for (int dt = 0; dt < 4; ++dt)
#pragma unroll
    for (int r = 0; r < 8; ++r) {
      const int qrow = q0 + half * 8 + r;
      const int d = dt * 16 + lr;
      const float v = ctx[dt][r] / lrow[r];
      ctxbuf[(size_t)(b * SEQ + qrow) * DMODEL + h * DK + d] = (__bf16)v;
    }
}

// ---------------------------------------------------------------- output projection
__global__ __launch_bounds__(32)
void out_proj_kernel(const __bf16* __restrict__ Ctx, const __bf16* __restrict__ Wo,
                     const float* __restrict__ bo, const float* __restrict__ resid,
                     float* __restrict__ preln) {
  const int lane = threadIdx.x;
  const int lr = lane & 15, half = lane >> 4;
  const int m0 = blockIdx.x * 64;
  const int n0 = blockIdx.y * 64;

  f32x8 c[4][4] = {};
  for (int k0 = 0; k0 < DMODEL; k0 += 32) {
    bf16x16 a[4];
#pragma unroll
    for (int mi = 0; mi < 4; ++mi)
      a[mi] = load_a_frag(Ctx + (m0 + mi * 16) * DMODEL + k0, DMODEL, lr, half);
#pragma unroll
    for (int ni = 0; ni < 4; ++ni) {
      bf16x16 bfr = load_bt_frag(Wo + (n0 + ni * 16) * DMODEL + k0, DMODEL, lr, half);
#pragma unroll
      for (int mi = 0; mi < 4; ++mi) c[mi][ni] = wmma_bf16f32(a[mi], bfr, c[mi][ni]);
    }
  }

#pragma unroll
  for (int mi = 0; mi < 4; ++mi)
#pragma unroll
    for (int ni = 0; ni < 4; ++ni) {
      const int n = n0 + ni * 16 + lr;
      const float bn = bo[n];
#pragma unroll
      for (int r = 0; r < 8; ++r) {
        const int m = m0 + mi * 16 + half * 8 + r;
        preln[m * DMODEL + n] = c[mi][ni][r] + bn + resid[m * DMODEL + n];
      }
    }
}

// ---------------------------------------------------------------- LayerNorm
__global__ __launch_bounds__(256)
void ln_kernel(const float* __restrict__ X, const float* __restrict__ gamma,
               const float* __restrict__ beta, float* __restrict__ Out) {
  __shared__ float red[256];
  const int row = blockIdx.x, tid = threadIdx.x;
  const float* x = X + (size_t)row * DMODEL;

  float v[4], s = 0.0f, sq = 0.0f;
#pragma unroll
  for (int i = 0; i < 4; ++i) {
    v[i] = x[tid + i * 256];
    s += v[i];
    sq += v[i] * v[i];
  }
  red[tid] = s;
  __syncthreads();
  for (int off = 128; off > 0; off >>= 1) {
    if (tid < off) red[tid] += red[tid + off];
    __syncthreads();
  }
  const float mean = red[0] * (1.0f / DMODEL);
  __syncthreads();
  red[tid] = sq;
  __syncthreads();
  for (int off = 128; off > 0; off >>= 1) {
    if (tid < off) red[tid] += red[tid + off];
    __syncthreads();
  }
  const float var = red[0] * (1.0f / DMODEL) - mean * mean;
  const float rstd = rsqrtf(var + 1e-5f);
#pragma unroll
  for (int i = 0; i < 4; ++i) {
    const int col = tid + i * 256;
    Out[(size_t)row * DMODEL + col] = (v[i] - mean) * rstd * gamma[col] + beta[col];
  }
}

// ---------------------------------------------------------------- host launcher
extern "C" void kernel_launch(void* const* d_in, const int* in_sizes, int n_in,
                              void* d_out, int out_size, void* d_ws, size_t ws_size,
                              hipStream_t stream) {
  const float* q = (const float*)d_in[0];
  const float* k = (const float*)d_in[1];
  const float* v = (const float*)d_in[2];
  const float* Wq = (const float*)d_in[3];
  const float* bq = (const float*)d_in[4];
  const float* Wk = (const float*)d_in[5];
  const float* bk = (const float*)d_in[6];
  const float* Wv = (const float*)d_in[7];
  const float* bv = (const float*)d_in[8];
  const float* Wo = (const float*)d_in[9];
  const float* bo = (const float*)d_in[10];
  const float* gamma = (const float*)d_in[11];
  const float* beta = (const float*)d_in[12];
  float* out = (float*)d_out;

  char* ws = (char*)d_ws;
  size_t off = 0;
  auto alloc = [&](size_t bytes) -> void* {
    void* p = ws + off;
    off += (bytes + 255) & ~(size_t)255;
    return p;
  };
  const size_t NX = (size_t)NROWS * DMODEL;   // 4M elements
  const size_t NW = (size_t)DMODEL * DMODEL;  // 1M elements

  __bf16* xq = (__bf16*)alloc(NX * 2);
  __bf16* xk = (__bf16*)alloc(NX * 2);
  __bf16* xv = (__bf16*)alloc(NX * 2);
  __bf16* wq = (__bf16*)alloc(NW * 2);
  __bf16* wk = (__bf16*)alloc(NW * 2);
  __bf16* wv = (__bf16*)alloc(NW * 2);
  __bf16* wo = (__bf16*)alloc(NW * 2);
  __bf16* Qhp = (__bf16*)alloc(NX * 2);
  __bf16* Khp = (__bf16*)alloc(NX * 2);
  __bf16* Vtp = (__bf16*)alloc(NX * 2);
  float* resid = (float*)alloc(NX * 4);
  __bf16* ctxb = (__bf16*)alloc(NX * 2);
  float* preln = (float*)alloc(NX * 4);

  cvt_f32_bf16<<<(int)(NX / 256), 256, 0, stream>>>(q, xq, (int)NX);
  cvt_f32_bf16<<<(int)(NX / 256), 256, 0, stream>>>(k, xk, (int)NX);
  cvt_f32_bf16<<<(int)(NX / 256), 256, 0, stream>>>(v, xv, (int)NX);
  cvt_f32_bf16<<<(int)(NW / 256), 256, 0, stream>>>(Wq, wq, (int)NW);
  cvt_f32_bf16<<<(int)(NW / 256), 256, 0, stream>>>(Wk, wk, (int)NW);
  cvt_f32_bf16<<<(int)(NW / 256), 256, 0, stream>>>(Wv, wv, (int)NW);
  cvt_f32_bf16<<<(int)(NW / 256), 256, 0, stream>>>(Wo, wo, (int)NW);

  qkv_proj_kernel<<<dim3(NROWS / 64, DMODEL / 64, 3), 32, 0, stream>>>(
      xq, xk, xv, wq, wk, wv, bq, bk, bv, resid, Qhp, Khp, Vtp);

  attn_kernel<<<dim3(SEQ / 64, NB * NHEADS), 128, 0, stream>>>(Qhp, Khp, Vtp, ctxb);

  out_proj_kernel<<<dim3(NROWS / 64, DMODEL / 64), 32, 0, stream>>>(ctxb, wo, bo,
                                                                    resid, preln);

  ln_kernel<<<NROWS, 256, 0, stream>>>(preln, gamma, beta, out);
}
